// Attention_3238405341737
// MI455X (gfx1250) — matrix operand order repaired and checked by reading, not compiled, not run on hardware
//
#include <hip/hip_runtime.h>

#define BB  2
#define SS  2048
#define DD  1024
#define HH  16
#define KVHN 4
#define HDN 64
#define MM  (BB * SS)   // 4096 rows

typedef _Float16 v16h __attribute__((ext_vector_type(16)));
typedef float    v8f  __attribute__((ext_vector_type(8)));

#define WMMA_F16(a, b, c) \
    __builtin_amdgcn_wmma_f32_16x16x32_f16(false, (a), false, (b), (short)0, (c), false, false)

// ---------- cross-lane row reductions within 16-lane halves (wave32) ----------
__device__ __forceinline__ float red_max16(float v) {
    v = fmaxf(v, __shfl_xor(v, 1, 32));
    v = fmaxf(v, __shfl_xor(v, 2, 32));
    v = fmaxf(v, __shfl_xor(v, 4, 32));
    v = fmaxf(v, __shfl_xor(v, 8, 32));
    return v;
}
__device__ __forceinline__ float red_sum16(float v) {
    v += __shfl_xor(v, 1, 32);
    v += __shfl_xor(v, 2, 32);
    v += __shfl_xor(v, 4, 32);
    v += __shfl_xor(v, 8, 32);
    return v;
}

// ---------- elementwise f32 -> f16 ----------
__global__ void cvt_f16(const float* __restrict__ src, _Float16* __restrict__ dst, int n) {
    int i = blockIdx.x * blockDim.x + threadIdx.x;
    if (i < n) dst[i] = (_Float16)src[i];
}

// ---------- transpose + convert: src[K,N] f32 -> dst[N,K] f16 ----------
__global__ void tcvt_f16(const float* __restrict__ src, _Float16* __restrict__ dst, int K, int N) {
    int i = blockIdx.x * blockDim.x + threadIdx.x;
    if (i < K * N) {
        int k = i / N, n = i - k * N;
        dst[(size_t)n * K + k] = (_Float16)src[i];
    }
}

// ---------- WMMA GEMM: C[M,N](f32) = A[M,K](f16, row-major) x Bt[N,K](f16) ----------
// one wave per 32x32 C tile: 2 A-frags x 2 B-frags -> 4 WMMAs per 32-wide k-step
__global__ __launch_bounds__(128) void wmma_gemm(const _Float16* __restrict__ A,
                                                 const _Float16* __restrict__ Bt,
                                                 float* __restrict__ C,
                                                 int M, int N, int K) {
    const int lane = threadIdx.x & 31;
    const int wid  = threadIdx.x >> 5;
    const int hl   = lane >> 4;       // lane half (0: lanes 0-15, 1: lanes 16-31)
    const int l15  = lane & 15;
    int tile = blockIdx.x * 4 + wid;
    int ntn  = N >> 5;
    int mt = tile / ntn, nt = tile - mt * ntn;
    if (mt >= (M >> 5)) return;
    const int m0 = mt << 5, n0 = nt << 5;

    const _Float16* arow0 = A  + (size_t)(m0 + l15) * K + 16 * hl;
    const _Float16* arow1 = arow0 + (size_t)16 * K;
    const _Float16* bcol0 = Bt + (size_t)(n0 + l15) * K + 16 * hl;
    const _Float16* bcol1 = bcol0 + (size_t)16 * K;

    v8f acc00 = {}, acc01 = {}, acc10 = {}, acc11 = {};
    for (int k0 = 0; k0 < K; k0 += 32) {
        // prefetch next k-chunk into near cache (global_prefetch_b8)
        if (k0 + 512 < K) {
            __builtin_prefetch(arow0 + k0 + 512, 0, 1);
            __builtin_prefetch(bcol0 + k0 + 512, 0, 1);
        }
        v16h a0, a1, b0, b1;
#pragma unroll
        for (int j = 0; j < 8; ++j) {
            a0[j] = arow0[k0 + j]; a0[j + 8] = arow0[k0 + 8 + j];
            a1[j] = arow1[k0 + j]; a1[j + 8] = arow1[k0 + 8 + j];
        }
#pragma unroll
        for (int j = 0; j < 16; ++j) { b0[j] = bcol0[k0 + j]; b1[j] = bcol1[k0 + j]; }
        acc00 = WMMA_F16(a0, b0, acc00);
        acc01 = WMMA_F16(a0, b1, acc01);
        acc10 = WMMA_F16(a1, b0, acc10);
        acc11 = WMMA_F16(a1, b1, acc11);
    }
    float* cb = C + (size_t)(m0 + 8 * hl) * N + n0 + l15;
#pragma unroll
    for (int r = 0; r < 8; ++r) {
        cb[(size_t)r * N]             = acc00[r];
        cb[(size_t)r * N + 16]        = acc01[r];
        cb[(size_t)(r + 16) * N]      = acc10[r];
        cb[(size_t)(r + 16) * N + 16] = acc11[r];
    }
}

// ---------- RoPE Q: raw[B,S,H*64] f32 -> qh[B,H,S,64] f16 ----------
__global__ void rope_q(const float* __restrict__ qr, const float* __restrict__ fc,
                       const float* __restrict__ fs, _Float16* __restrict__ qh) {
    int i = blockIdx.x * blockDim.x + threadIdx.x;
    if (i >= BB * SS * HH * 32) return;
    int p = i & 31;  int t = i >> 5;
    int h = t & 15;  t >>= 4;
    int s = t & (SS - 1); int b = t >> 11;
    float c  = fc[s * 32 + p], sn = fs[s * 32 + p];
    size_t ib = ((size_t)(b * SS + s)) * DD + h * HDN + 2 * p;
    float tr = qr[ib], ti = qr[ib + 1];
    size_t ob = (((size_t)(b * HH + h)) * SS + s) * HDN + 2 * p;
    qh[ob]     = (_Float16)(tr * c - ti * sn);
    qh[ob + 1] = (_Float16)(tr * sn + ti * c);
}

// ---------- RoPE K: raw[B,S,4*64] f32 -> kh[B,4,S,64] f16 ----------
__global__ void rope_k(const float* __restrict__ kr, const float* __restrict__ fc,
                       const float* __restrict__ fs, _Float16* __restrict__ kh) {
    int i = blockIdx.x * blockDim.x + threadIdx.x;
    if (i >= BB * SS * KVHN * 32) return;
    int p = i & 31;  int t = i >> 5;
    int kv = t & 3;  t >>= 2;
    int s = t & (SS - 1); int b = t >> 11;
    float c  = fc[s * 32 + p], sn = fs[s * 32 + p];
    size_t ib = ((size_t)(b * SS + s)) * (KVHN * HDN) + kv * HDN + 2 * p;
    float tr = kr[ib], ti = kr[ib + 1];
    size_t ob = (((size_t)(b * KVHN + kv)) * SS + s) * HDN + 2 * p;
    kh[ob]     = (_Float16)(tr * c - ti * sn);
    kh[ob + 1] = (_Float16)(tr * sn + ti * c);
}

// ---------- V pack: raw[B,S,4*64] f32 -> vt[B,4,64,S] f16 (transposed) ----------
__global__ void pack_v(const float* __restrict__ vr, _Float16* __restrict__ vt) {
    int i = blockIdx.x * blockDim.x + threadIdx.x;
    if (i >= BB * SS * KVHN * HDN) return;
    int d = i & 63;  int t = i >> 6;
    int kv = t & 3;  t >>= 2;
    int s = t & (SS - 1); int b = t >> 11;
    float v = vr[((size_t)(b * SS + s)) * (KVHN * HDN) + kv * HDN + d];
    vt[(((size_t)(b * KVHN + kv)) * HDN + d) * SS + s] = (_Float16)v;
}

// ---------- Flash attention: wave per 16-row Q tile, online softmax ----------
// Qh[B,H,S,64], Kh[B,4,S,64], Vt[B,4,64,S] -> Attn[B,S,H*64] f16
__global__ __launch_bounds__(128) void flash_attn(const _Float16* __restrict__ Qh,
                                                  const _Float16* __restrict__ Kh,
                                                  const _Float16* __restrict__ Vt,
                                                  _Float16* __restrict__ Attn) {
    __shared__ _Float16 Pt[4][16 * 32];   // per-wave P staging (C-layout -> A-layout)
    const int lane = threadIdx.x & 31;
    const int wid  = threadIdx.x >> 5;
    const int hl   = lane >> 4;
    const int l15  = lane & 15;
    int gid = blockIdx.x * 4 + wid;       // 0 .. B*H*(S/16)-1
    int qt  = gid & (SS / 16 - 1);
    int bh  = gid >> 7;
    int h   = bh & (HH - 1);
    int b   = bh >> 4;
    int kvh = h >> 2;                      // REP = 4
    const int q0 = qt << 4;

    const _Float16* qb = Qh + (((size_t)(b * HH + h)) * SS + q0) * HDN;
    const _Float16* kb = Kh + ((size_t)(b * KVHN + kvh)) * SS * HDN;
    const _Float16* vb = Vt + ((size_t)(b * KVHN + kvh)) * HDN * SS;

    // Q fragments: K = 0..31 and 32..63
    v16h aq0, aq1;
    {
        const _Float16* ar = qb + (size_t)l15 * HDN + 16 * hl;
#pragma unroll
        for (int j = 0; j < 8; ++j) {
            aq0[j] = ar[j];        aq0[j + 8] = ar[8 + j];
            aq1[j] = ar[32 + j];   aq1[j + 8] = ar[40 + j];
        }
    }

    v8f oacc[4] = {{}, {}, {}, {}};
    float mx[8], sm[8];
#pragma unroll
    for (int r = 0; r < 8; ++r) { mx[r] = -3.0e38f; sm[r] = 0.f; }

    _Float16* P = Pt[wid];
    const int nsub = qt + 1;               // # of valid 16-key subtiles (causal)
    for (int p = 0; 2 * p < nsub; ++p) {
        const int ks0 = p * 32;
        const bool v1 = (2 * p + 1) < nsub;
        v8f s0 = {}, s1 = {};
        {   // score subtile 0: keys ks0..ks0+15
            const _Float16* br = kb + (size_t)(ks0 + l15) * HDN + 16 * hl;
            v16h b0, b1;
#pragma unroll
            for (int j = 0; j < 16; ++j) { b0[j] = br[j]; b1[j] = br[32 + j]; }
            s0 = WMMA_F16(aq0, b0, s0);
            s0 = WMMA_F16(aq1, b1, s0);
        }
        if (v1) {  // score subtile 1: keys ks0+16..ks0+31
            const _Float16* br = kb + (size_t)(ks0 + 16 + l15) * HDN + 16 * hl;
            v16h b0, b1;
#pragma unroll
            for (int j = 0; j < 16; ++j) { b0[j] = br[j]; b1[j] = br[32 + j]; }
            s1 = WMMA_F16(aq0, b0, s1);
            s1 = WMMA_F16(aq1, b1, s1);
        }
        // scale + causal mask + online softmax update (fp32)
#pragma unroll
        for (int r = 0; r < 8; ++r) {
            int row = q0 + r + 8 * hl;
            int c0 = ks0 + l15, c1 = ks0 + 16 + l15;
            float u0 = s0[r] * 0.125f + (c0 <= row ? 0.f : -1.0e9f);
            float u1 = v1 ? (s1[r] * 0.125f + (c1 <= row ? 0.f : -1.0e9f)) : -1.0e9f;
            float gm = red_max16(fmaxf(u0, u1));
            float nm = fmaxf(mx[r], gm);
            float corr = __expf(mx[r] - nm);
            mx[r] = nm;
            float e0 = __expf(u0 - nm);
            float e1 = __expf(u1 - nm);
            sm[r] = sm[r] * corr + red_sum16(e0 + e1);
#pragma unroll
            for (int c = 0; c < 4; ++c) oacc[c][r] *= corr;
            P[(r + 8 * hl) * 32 + l15]      = (_Float16)e0;
            P[(r + 8 * hl) * 32 + 16 + l15] = (_Float16)e1;
        }
        // reload P as A fragment (16x32 f16); same-wave LDS ops are in-order
        v16h ap;
#pragma unroll
        for (int j = 0; j < 8; ++j) {
            ap[j]     = P[l15 * 32 + 16 * hl + j];
            ap[j + 8] = P[l15 * 32 + 16 * hl + 8 + j];
        }
        // PV: O[16x64] += P[16x32] x V[32x64]  (4 n-chunks of 16)
#pragma unroll
        for (int c = 0; c < 4; ++c) {
            const _Float16* vr = vb + (size_t)(c * 16 + l15) * SS;
            v16h bv;
            if (v1) {
                const _Float16* vp = vr + ks0 + 16 * hl;
#pragma unroll
                for (int j = 0; j < 16; ++j) bv[j] = vp[j];
            } else {
#pragma unroll
                for (int j = 0; j < 16; ++j) {
                    int kk = ks0 + 16 * hl + j;
                    if (kk > SS - 1) kk = SS - 1;   // masked keys: P == 0
                    bv[j] = vr[kk];
                }
            }
            oacc[c] = WMMA_F16(ap, bv, oacc[c]);
        }
    }
    // normalize + store f16 to Attn[B,S,H*64]
#pragma unroll
    for (int r = 0; r < 8; ++r) {
        float inv = 1.0f / sm[r];
        size_t ro = ((size_t)b * SS + q0 + r + 8 * hl) * DD + h * HDN + l15;
        Attn[ro]      = (_Float16)(oacc[0][r] * inv);
        Attn[ro + 16] = (_Float16)(oacc[1][r] * inv);
        Attn[ro + 32] = (_Float16)(oacc[2][r] * inv);
        Attn[ro + 48] = (_Float16)(oacc[3][r] * inv);
    }
}

extern "C" void kernel_launch(void* const* d_in, const int* in_sizes, int n_in,
                              void* d_out, int out_size, void* d_ws, size_t ws_size,
                              hipStream_t stream) {
    (void)in_sizes; (void)n_in; (void)out_size; (void)ws_size;
    const float* x  = (const float*)d_in[0];
    const float* wq = (const float*)d_in[1];
    const float* wk = (const float*)d_in[2];
    const float* wv = (const float*)d_in[3];
    const float* wo = (const float*)d_in[4];
    const float* fc = (const float*)d_in[5];
    const float* fs = (const float*)d_in[6];
    // d_in[7] (mask) unused: causal mask computed inline
    float* out = (float*)d_out;

    char* ws = (char*)d_ws;
    size_t off = 0;
    auto alloc = [&](size_t bytes) -> void* {
        void* p = ws + off;
        off += (bytes + 255) & ~(size_t)255;
        return p;
    };
    _Float16* xh   = (_Float16*)alloc((size_t)MM * DD * 2);
    _Float16* wqT  = (_Float16*)alloc((size_t)DD * DD * 2);
    _Float16* wkT  = (_Float16*)alloc((size_t)256 * DD * 2);
    _Float16* wvT  = (_Float16*)alloc((size_t)256 * DD * 2);
    _Float16* woT  = (_Float16*)alloc((size_t)DD * DD * 2);
    float*    qraw = (float*)   alloc((size_t)MM * DD * 4);
    float*    kraw = (float*)   alloc((size_t)MM * 256 * 4);
    float*    vraw = (float*)   alloc((size_t)MM * 256 * 4);
    _Float16* qh   = (_Float16*)alloc((size_t)MM * DD * 2);
    _Float16* kh   = (_Float16*)alloc((size_t)MM * 256 * 2);
    _Float16* vt   = (_Float16*)alloc((size_t)MM * 256 * 2);
    _Float16* attn = (_Float16*)alloc((size_t)MM * DD * 2);

    const int T = 256;
    cvt_f16 <<<(MM * DD + T - 1) / T, T, 0, stream>>>(x, xh, MM * DD);
    tcvt_f16<<<(DD * DD + T - 1) / T, T, 0, stream>>>(wq, wqT, DD, DD);
    tcvt_f16<<<(DD * 256 + T - 1) / T, T, 0, stream>>>(wk, wkT, DD, 256);
    tcvt_f16<<<(DD * 256 + T - 1) / T, T, 0, stream>>>(wv, wvT, DD, 256);
    tcvt_f16<<<(DD * DD + T - 1) / T, T, 0, stream>>>(wo, woT, DD, DD);

    // QKV projections (f16 WMMA, fp32 accumulate), 32x32 tile per wave
    wmma_gemm<<<((MM / 32) * (DD / 32) + 3) / 4, 128, 0, stream>>>(xh, wqT, qraw, MM, DD, DD);
    wmma_gemm<<<((MM / 32) * (256 / 32) + 3) / 4, 128, 0, stream>>>(xh, wkT, kraw, MM, 256, DD);
    wmma_gemm<<<((MM / 32) * (256 / 32) + 3) / 4, 128, 0, stream>>>(xh, wvT, vraw, MM, 256, DD);

    rope_q<<<(BB * SS * HH * 32 + T - 1) / T, T, 0, stream>>>(qraw, fc, fs, qh);
    rope_k<<<(BB * SS * KVHN * 32 + T - 1) / T, T, 0, stream>>>(kraw, fc, fs, kh);
    pack_v<<<(BB * SS * KVHN * HDN + T - 1) / T, T, 0, stream>>>(vraw, vt);

    // flash attention: B*H*(S/16) = 4096 waves, 4 waves/block
    flash_attn<<<BB * HH * (SS / 16) / 4, 128, 0, stream>>>(qh, kh, vt, attn);

    // output projection straight into d_out (fp32)
    wmma_gemm<<<((MM / 32) * (DD / 32) + 3) / 4, 128, 0, stream>>>(attn, woT, out, MM, DD, DD);
}
